// SelfAttentionHybrid_65481071408014
// MI455X (gfx1250) — matrix operand order repaired and checked by reading, not compiled, not run on hardware
//
#include <hip/hip_runtime.h>
#include <hip/hip_bf16.h>

// SelfAttention (single head, E=1024) for MI455X gfx1250.
// All matmuls via v_wmma_f32_16x16x32_f16 (f16 in, f32 accum).
// Compute-bound workload (~137 GFLOP vs ~100MB essential traffic):
// unfused pipeline, intermediates live in the 192MB L2.
// Wave tile 64x64 (4x4 WMMA subtiles): 16 WMMA per 8 fragment loads
// = 32 FLOP per byte loaded from the memory hierarchy.

#define EMBED 1024
#define BATCH 4
#define SEQ   2048
#define MTOT  (BATCH * SEQ)   // 8192 rows

typedef __attribute__((ext_vector_type(16))) _Float16 v16h;
typedef __attribute__((ext_vector_type(8)))  _Float16 v8h;
typedef __attribute__((ext_vector_type(8)))  float    v8f;

// ---------------------------------------------------------------------------
// f32 -> f16 conversion (grid-stride)
// ---------------------------------------------------------------------------
__global__ void cvt_f32_to_f16(const float* __restrict__ in,
                               _Float16* __restrict__ out, int n) {
  int i = blockIdx.x * blockDim.x + threadIdx.x;
  int stride = gridDim.x * blockDim.x;
  for (; i < n; i += stride) out[i] = (_Float16)in[i];
}

// ---------------------------------------------------------------------------
// WMMA fragment loaders (layouts per CDNA5 ISA 7.12.2, wave32)
// ---------------------------------------------------------------------------
// A-matrix 16x32 f16: lane holds row M=lane&15.
//   halfs 0..7  -> K = (lane>>4)*8 + 0..7
//   halfs 8..15 -> K = 16 + (lane>>4)*8 + 0..7
__device__ inline v16h load_frag_a(const _Float16* __restrict__ A, int lda,
                                   int m0, int k0, int lane) {
  int r  = m0 + (lane & 15);
  int kb = k0 + ((lane >> 4) << 3);
  const _Float16* p = A + (long)r * lda + kb;
  v8h lo = *(const v8h*)p;
  v8h hi = *(const v8h*)(p + 16);
  return __builtin_shufflevector(lo, hi, 0, 1, 2, 3, 4, 5, 6, 7,
                                 8, 9, 10, 11, 12, 13, 14, 15);
}

// B-matrix 32x16 f16, sourced from a row-major [N, K] array (i.e. B = W^T):
//   lane holds column N=lane&15; halfs 0..7 -> K=(lane>>4)*8+0..7, halfs 8..15 -> K=16+(lane>>4)*8+0..7.
__device__ inline v16h load_frag_b(const _Float16* __restrict__ Bm, int ldb,
                                   int n0, int k0, int lane) {
  int n  = n0 + (lane & 15);
  int kb = k0 + ((lane >> 4) << 3);
  const _Float16* p = Bm + (long)n * ldb + kb;
  v8h lo = *(const v8h*)p;
  v8h hi = *(const v8h*)(p + 16);
  return __builtin_shufflevector(lo, hi, 0, 1, 2, 3, 4, 5, 6, 7,
                                 8, 9, 10, 11, 12, 13, 14, 15);
}

// ---------------------------------------------------------------------------
// Generic WMMA GEMM:  C[M,N] = scale * (A16[M,K] @ Wrow[N,K]^T) + bias[N]
// Block: 256 threads = 8 waves arranged 2(M) x 4(N); wave tile 64x64
// (4x4 WMMA subtiles); WG tile 128x256.
// mode 0: store f16 row-major C16[row*ldc+col]
// mode 1: store f16 transposed-per-batch: vT[b][col][s], b=row/SEQ (V proj)
// mode 2: store f32 row-major C32[row*ldc+col]
// gridDim.z = batch; A/B/C advanced by strideA/B/C elements per z.
// All dims assumed multiples of the tiling (true for every launch here).
// ---------------------------------------------------------------------------
__global__ __launch_bounds__(256)
void gemm_wmma_f16(const _Float16* __restrict__ A,
                   const _Float16* __restrict__ Bm,
                   const float* __restrict__ bias,
                   _Float16* __restrict__ C16,
                   float* __restrict__ C32,
                   int K, int lda, int ldb, int ldc,
                   long strideA, long strideB, long strideC,
                   float scale, int mode) {
  const int z = blockIdx.z;
  A  += (long)z * strideA;
  Bm += (long)z * strideB;

  const int lane = threadIdx.x & 31;
  const int wave = threadIdx.x >> 5;
  const int m0 = blockIdx.x * 128 + (wave >> 2) * 64;   // 2 waves in M
  const int n0 = blockIdx.y * 256 + (wave & 3) * 64;    // 4 waves in N

  v8f acc[4][4];
#pragma unroll
  for (int i = 0; i < 4; ++i)
#pragma unroll
    for (int j = 0; j < 4; ++j) acc[i][j] = (v8f){};

  for (int k0 = 0; k0 < K; k0 += 32) {
    v16h a[4], b[4];
#pragma unroll
    for (int i = 0; i < 4; ++i) a[i] = load_frag_a(A, lda, m0 + 16 * i, k0, lane);
#pragma unroll
    for (int j = 0; j < 4; ++j) b[j] = load_frag_b(Bm, ldb, n0 + 16 * j, k0, lane);
#pragma unroll
    for (int i = 0; i < 4; ++i)
#pragma unroll
      for (int j = 0; j < 4; ++j)
        acc[i][j] = __builtin_amdgcn_wmma_f32_16x16x32_f16(
            false, a[i], false, b[j], (short)0, acc[i][j], false, false);

    // L2 prefetch of the streaming A operand (emits global_prefetch_b8).
    __builtin_prefetch(A + (long)(m0 + (lane & 15)) * lda + (k0 + 256), 0, 1);
  }

  // Store one 16x16 tile. C/D layout: VGPR r -> (M = r + (lane>>4)*8, N = lane&15)
  if (mode == 0) C16 += (long)z * strideC;
  const int colOff = lane & 15;
  const int rOff   = (lane >> 4) << 3;
#pragma unroll
  for (int i = 0; i < 4; ++i) {
#pragma unroll
    for (int j = 0; j < 4; ++j) {
      const int col   = n0 + 16 * j + colOff;
      const int rbase = m0 + 16 * i + rOff;
      const float bv  = bias ? bias[col] : 0.0f;
      if (mode == 1) {
        // V projection -> vT[b][e][s]; a lane's 8 acc elements are 8
        // consecutive s positions -> single 16B store.
        int b = rbase >> 11;                 // SEQ = 2048
        int s = rbase & (SEQ - 1);
        v8h h;
#pragma unroll
        for (int r = 0; r < 8; ++r) h[r] = (_Float16)(acc[i][j][r] * scale + bv);
        *(v8h*)(C16 + (long)b * (EMBED * (long)SEQ) + (long)col * SEQ + s) = h;
      } else {
#pragma unroll
        for (int r = 0; r < 8; ++r) {
          float v = acc[i][j][r] * scale + bv;
          int row = rbase + r;
          if (mode == 0) C16[(long)row * ldc + col] = (_Float16)v;
          else           C32[(long)z * strideC + (long)row * ldc + col] = v;
        }
      }
    }
  }
}

// ---------------------------------------------------------------------------
// Row softmax: one 256-thread block per row of 2048 f32 scores -> f16 probs.
// ---------------------------------------------------------------------------
__global__ __launch_bounds__(256)
void softmax_rows(const float* __restrict__ S, _Float16* __restrict__ P) {
  const long row = blockIdx.x;
  const float* srow = S + row * SEQ;
  _Float16* prow = P + row * SEQ;
  const int tid = threadIdx.x;

  float v[8];
  float m = -3.4e38f;
#pragma unroll
  for (int t = 0; t < 8; ++t) {
    v[t] = srow[tid + 256 * t];
    m = fmaxf(m, v[t]);
  }
#pragma unroll
  for (int off = 16; off >= 1; off >>= 1)
    m = fmaxf(m, __shfl_xor(m, off, 32));

  __shared__ float red[8];
  __shared__ float bc;
  if ((tid & 31) == 0) red[tid >> 5] = m;
  __syncthreads();
  if (tid == 0) {
    float mm = red[0];
#pragma unroll
    for (int i = 1; i < 8; ++i) mm = fmaxf(mm, red[i]);
    bc = mm;
  }
  __syncthreads();
  m = bc;

  float s = 0.0f;
#pragma unroll
  for (int t = 0; t < 8; ++t) {
    v[t] = __expf(v[t] - m);
    s += v[t];
  }
#pragma unroll
  for (int off = 16; off >= 1; off >>= 1)
    s += __shfl_xor(s, off, 32);
  __syncthreads();                       // red[] reuse
  if ((tid & 31) == 0) red[tid >> 5] = s;
  __syncthreads();
  if (tid == 0) {
    float ss = 0.0f;
#pragma unroll
    for (int i = 0; i < 8; ++i) ss += red[i];
    bc = ss;
  }
  __syncthreads();
  const float inv = 1.0f / bc;
#pragma unroll
  for (int t = 0; t < 8; ++t)
    prow[tid + 256 * t] = (_Float16)(v[t] * inv);
}

// ---------------------------------------------------------------------------
// Host-side orchestration
// ---------------------------------------------------------------------------
extern "C" void kernel_launch(void* const* d_in, const int* in_sizes, int n_in,
                              void* d_out, int out_size, void* d_ws, size_t ws_size,
                              hipStream_t stream) {
  const float* x  = (const float*)d_in[0];
  const float* Wq = (const float*)d_in[1];
  const float* bq = (const float*)d_in[2];
  const float* Wk = (const float*)d_in[3];
  const float* bk = (const float*)d_in[4];
  const float* Wv = (const float*)d_in[5];
  const float* bv = (const float*)d_in[6];
  const float* Wo = (const float*)d_in[7];
  const float* bo = (const float*)d_in[8];
  float* out = (float*)d_out;

  const size_t MB = 1u << 20;
  char* ws = (char*)d_ws;
  // Workspace map (136 MB, with dead-buffer aliasing):
  _Float16* x16    = (_Float16*)(ws);              // 16 MB  (dead after projections)
  _Float16* wq16   = (_Float16*)(ws + 16 * MB);    //  2 MB
  _Float16* wk16   = (_Float16*)(ws + 18 * MB);    //  2 MB
  _Float16* wv16   = (_Float16*)(ws + 20 * MB);    //  2 MB
  _Float16* wo16   = (_Float16*)(ws + 22 * MB);    //  2 MB
  _Float16* q16    = (_Float16*)(ws + 24 * MB);    // 16 MB  (dead after scores)
  _Float16* k16    = (_Float16*)(ws + 40 * MB);    // 16 MB  (dead after scores)
  _Float16* vT16   = (_Float16*)(ws + 56 * MB);    // 16 MB  [B][E][S]
  float*    scores = (float*)   (ws + 72 * MB);    // 64 MB  [B][S][S]
  _Float16* attn16 = (_Float16*)(ws + 24 * MB);    // 32 MB  overlays q16+k16
  _Float16* out16  = (_Float16*)(ws);              // 16 MB  overlays x16

  dim3 blk(256);

  // 1) precision conversion
  cvt_f32_to_f16<<<4096, blk, 0, stream>>>(x,  x16,  MTOT * EMBED);
  cvt_f32_to_f16<<<1024, blk, 0, stream>>>(Wq, wq16, EMBED * EMBED);
  cvt_f32_to_f16<<<1024, blk, 0, stream>>>(Wk, wk16, EMBED * EMBED);
  cvt_f32_to_f16<<<1024, blk, 0, stream>>>(Wv, wv16, EMBED * EMBED);
  cvt_f32_to_f16<<<1024, blk, 0, stream>>>(Wo, wo16, EMBED * EMBED);

  // 2) Q/K/V projections: [8192,1024] x [1024,1024]^T (+bias)
  dim3 gproj(MTOT / 128, EMBED / 256, 1);
  gemm_wmma_f16<<<gproj, blk, 0, stream>>>(x16, wq16, bq, q16, nullptr,
      EMBED, EMBED, EMBED, EMBED, 0, 0, 0, 1.0f, /*mode=*/0);
  gemm_wmma_f16<<<gproj, blk, 0, stream>>>(x16, wk16, bk, k16, nullptr,
      EMBED, EMBED, EMBED, EMBED, 0, 0, 0, 1.0f, /*mode=*/0);
  gemm_wmma_f16<<<gproj, blk, 0, stream>>>(x16, wv16, bv, vT16, nullptr,
      EMBED, EMBED, EMBED, EMBED, 0, 0, 0, 1.0f, /*mode=*/1);  // stores V^T

  // 3) scores = (Q @ K^T) / sqrt(E)   per batch, f32
  dim3 gsc(SEQ / 128, SEQ / 256, BATCH);
  gemm_wmma_f16<<<gsc, blk, 0, stream>>>(q16, k16, nullptr, nullptr, scores,
      EMBED, EMBED, EMBED, SEQ,
      (long)SEQ * EMBED, (long)SEQ * EMBED, (long)SEQ * SEQ,
      0.03125f /*1/sqrt(1024)*/, /*mode=*/2);

  // 4) softmax rows -> f16 probs (overlays q16/k16)
  softmax_rows<<<BATCH * SEQ, blk, 0, stream>>>(scores, attn16);

  // 5) out = attn @ V   (B is vT: row-major [E, S] per batch)
  dim3 gav(SEQ / 128, EMBED / 256, BATCH);
  gemm_wmma_f16<<<gav, blk, 0, stream>>>(attn16, vT16, nullptr, out16, nullptr,
      SEQ, SEQ, SEQ, EMBED,
      (long)SEQ * SEQ, (long)EMBED * SEQ, (long)SEQ * EMBED,
      1.0f, /*mode=*/0);

  // 6) final projection: d_out = out @ Wo^T + bo   (f32)
  gemm_wmma_f16<<<gproj, blk, 0, stream>>>(out16, wo16, bo, nullptr, out,
      EMBED, EMBED, EMBED, EMBED, 0, 0, 0, 1.0f, /*mode=*/2);
}
